// VQVAE_41300405518708
// MI455X (gfx1250) — matrix operand order repaired and checked by reading, not compiled, not run on hardware
//
#include <hip/hip_runtime.h>
#include <hip/hip_bf16.h>

typedef __attribute__((ext_vector_type(16))) _Float16 v16h;
typedef __attribute__((ext_vector_type(8)))  _Float16 v8h;
typedef __attribute__((ext_vector_type(8)))  float    v8f;

// 16 contiguous halves -> v16h (two b128 loads, no conversion).
__device__ __forceinline__ v16h load16h(const _Float16* __restrict__ p) {
  v8h a = *(const v8h*)p;
  v8h b = *(const v8h*)(p + 8);
  return __builtin_shufflevector(a, b, 0,1,2,3,4,5,6,7,8,9,10,11,12,13,14,15);
}

// A-tile: two 8-half runs at p and p+16 (ic base8 and base8+16), optional ReLU.
template<bool RELU>
__device__ __forceinline__ v16h load_a_runs(const _Float16* __restrict__ p) {
  v8h a = *(const v8h*)p;
  v8h b = *(const v8h*)(p + 16);
  v16h v = __builtin_shufflevector(a, b, 0,1,2,3,4,5,6,7,8,9,10,11,12,13,14,15);
  if (RELU) {
#pragma unroll
    for (int h = 0; h < 16; ++h)
      v[h] = (v[h] > (_Float16)0.f) ? v[h] : (_Float16)0.f;   // v_pk_max
  }
  return v;
}

// ---------------------------------------------------------------------------
// conv e1: x (32,1,256,256) f32 -> h1 f16 NHWC (32,128,128,32), k4 s2 p1,
// bias, ReLU.  One thread per pixel, uniform (scalar-cache) weight reads.
// ---------------------------------------------------------------------------
__global__ void conv_e1_kernel(const float* __restrict__ x,
                               const float* __restrict__ w,   // (32,1,4,4)
                               const float* __restrict__ bias,
                               _Float16* __restrict__ out) {
  int p = blockIdx.x * 256 + threadIdx.x;     // 524288 pixels
  int ox = p & 127; int t = p >> 7;
  int oy = t & 127; int b = t >> 7;
  float acc[32];
#pragma unroll
  for (int oc = 0; oc < 32; ++oc) acc[oc] = bias[oc];
  int iy0 = oy * 2 - 1, ix0 = ox * 2 - 1;
  const float* xb = x + ((size_t)b << 16);
#pragma unroll
  for (int ky = 0; ky < 4; ++ky) {
    int iy = iy0 + ky;
    if (iy < 0 || iy >= 256) continue;
#pragma unroll
    for (int kx = 0; kx < 4; ++kx) {
      int ix = ix0 + kx;
      if (ix < 0 || ix >= 256) continue;
      float v = xb[iy * 256 + ix];
#pragma unroll
      for (int oc = 0; oc < 32; ++oc)
        acc[oc] += v * w[oc * 16 + ky * 4 + kx];   // uniform -> s_load
    }
  }
  v8h* o = (v8h*)(out + (size_t)p * 32);
#pragma unroll
  for (int j = 0; j < 4; ++j) {
    v8h v;
#pragma unroll
    for (int e = 0; e < 8; ++e) v[e] = (_Float16)fmaxf(acc[8*j+e], 0.f);
    o[j] = v;
  }
}

// ---------------------------------------------------------------------------
// Repack OIHW f32 -> f16 [oc][(ky*KW+kx)*IC + ic] (tap-major K blocking).
// With KH=KW=1 this is a pure f32->f16 convert (also used for the codebook).
// ---------------------------------------------------------------------------
template<int IC, int KH, int KW>
__global__ void repack_oihw_kernel(const float* __restrict__ w,
                                   _Float16* __restrict__ wp, int total) {
  int i = blockIdx.x * 256 + threadIdx.x;
  if (i >= total) return;
  constexpr int K = IC * KH * KW;
  int oc = i / K;  int k  = i - oc * K;
  int kyx = k / IC; int ic = k - kyx * IC;
  wp[i] = (_Float16)w[(size_t)(oc * IC + ic) * (KH * KW) + kyx];
}

// ---------------------------------------------------------------------------
// f16 NHWC implicit-GEMM conv via v_wmma_f32_16x16x32_f16.
// A tile: 2x b128 (16 halves), one bounds check per 32-step; B tile: 2x b128.
// One wave computes MREP=2 16x16 tiles sharing the B registers.
// ---------------------------------------------------------------------------
template<int IC, int IH, int IW, int OC, int KH, int KW, int STRIDE, int PAD,
         bool RELU_IN, bool RELU_OUT, bool HAS_BIAS, bool HAS_RES>
__global__ __launch_bounds__(128) void wmma_conv_kernel(
    const _Float16* __restrict__ in,   // NHWC (B,IH,IW,IC)
    const _Float16* __restrict__ w,    // packed [oc][K]
    const float* __restrict__ bias, const _Float16* __restrict__ resid,
    _Float16* __restrict__ out) {      // NHWC (B,64,64,OC)
  constexpr int K      = IC * KH * KW;
  constexpr int NSTEP  = K / 32;
  constexpr int SUBS   = IC / 32;
  constexpr bool NOCHK = (KH == 1 && KW == 1 && PAD == 0);
  constexpr int MREP = 2;
  const int lane = threadIdx.x;
  const int col  = lane & 15;
  const int hi   = lane >> 4;
  const int job  = blockIdx.x * blockDim.y + threadIdx.y;
  const int n    = blockIdx.y * 16 + col;
  const _Float16* wn = w + (size_t)n * K + hi * 8;

  int iy0[MREP], ix0[MREP], bOff[MREP];
#pragma unroll
  for (int r = 0; r < MREP; ++r) {
    int m = (job * MREP + r) * 16 + col;
    int bat = m >> 12, rem = m & 4095;
    iy0[r] = (rem >> 6) * STRIDE - PAD;
    ix0[r] = (rem & 63) * STRIDE - PAD;
    bOff[r] = bat * (IH * IW);
  }

  v8f acc[MREP];
#pragma unroll
  for (int r = 0; r < MREP; ++r) acc[r] = (v8f){};

#pragma unroll
  for (int step = 0; step < NSTEP; ++step) {
    const int kyx = step / SUBS;             // constants after unroll
    const int s   = step - kyx * SUBS;
    const int ky  = kyx / KW, kx = kyx - (kyx / KW) * KW;
    v16h bv = load_a_runs<false>(wn + step * 32);
#pragma unroll
    for (int r = 0; r < MREP; ++r) {
      int iy = iy0[r] + ky, ix = ix0[r] + kx;
      v16h av;
      bool ok = NOCHK || (iy >= 0 && iy < IH && ix >= 0 && ix < IW);
      if (ok) {
        const _Float16* ap =
            in + (size_t)(bOff[r] + iy * IW + ix) * IC + s * 32 + hi * 8;
        av = load_a_runs<RELU_IN>(ap);
      } else {
#pragma unroll
        for (int h = 0; h < 16; ++h) av[h] = (_Float16)0.f;
      }
      acc[r] = __builtin_amdgcn_wmma_f32_16x16x32_f16(
          false, av, false, bv, (short)0, acc[r], false, false);
    }
  }

  const float bn = HAS_BIAS ? bias[n] : 0.f;
#pragma unroll
  for (int r = 0; r < MREP; ++r) {
#pragma unroll
    for (int rr = 0; rr < 8; ++rr) {
      int ms = (job * MREP + r) * 16 + rr + 8 * hi;
      size_t oidx = (size_t)ms * OC + n;        // NHWC: coalesced across lanes
      float v = acc[r][rr] + bn;
      if constexpr (HAS_RES) v += (float)resid[oidx];
      if constexpr (RELU_OUT) v = fmaxf(v, 0.f);
      out[oidx] = (_Float16)v;
    }
  }
}

// ---------------------------------------------------------------------------
// Codebook squared norms (512 x 64), f32.
// ---------------------------------------------------------------------------
__global__ void cbnorm_kernel(const float* __restrict__ cb,
                              float* __restrict__ norms) {
  int nn = blockIdx.x * 256 + threadIdx.x;
  if (nn >= 512) return;
  const float4* e = (const float4*)(cb + nn * 64);
  float s = 0.f;
#pragma unroll
  for (int d = 0; d < 16; ++d) {
    float4 v = e[d];
    s += v.x * v.x + v.y * v.y + v.z * v.z + v.w * v.w;
  }
  norms[nn] = s;
}

// ---------------------------------------------------------------------------
// VQ on f16 NHWC (32,64,64,64): A resident in regs (4x b128), 32 code tiles
// via WMMA against f16 codebook, f32 argmin of ||e||^2 - 2 x.e, then
// coalesced f16 codebook-row gather into q.
// ---------------------------------------------------------------------------
__global__ __launch_bounds__(128) void vq_kernel(
    const _Float16* __restrict__ xin, const _Float16* __restrict__ cbh,
    const float* __restrict__ cbn, _Float16* __restrict__ q) {
  const int lane = threadIdx.x;
  const int m_base = (blockIdx.x * blockDim.y + threadIdx.y) * 16;
  const int col = lane & 15;
  const int hi  = lane >> 4;

  const _Float16* xp = xin + (size_t)(m_base + col) * 64 + hi * 8;
  v16h a0 = load_a_runs<false>(xp);        // dims base8..+8, base8+16..+24
  v16h a1 = load_a_runs<false>(xp + 32);

  float bestv[8]; int besti[8];
#pragma unroll
  for (int r = 0; r < 8; ++r) { bestv[r] = 3.4e38f; besti[r] = 0; }

  for (int nt = 0; nt < 32; ++nt) {
    int nCol = nt * 16 + col;
    const _Float16* en = cbh + nCol * 64 + hi * 8;
    v16h b0 = load_a_runs<false>(en);
    v16h b1 = load_a_runs<false>(en + 32);
    v8f c = {};
    c = __builtin_amdgcn_wmma_f32_16x16x32_f16(false, a0, false, b0, (short)0, c, false, false);
    c = __builtin_amdgcn_wmma_f32_16x16x32_f16(false, a1, false, b1, (short)0, c, false, false);
    float nrm = cbn[nCol];
#pragma unroll
    for (int r = 0; r < 8; ++r) {
      float dval = nrm - 2.0f * c[r];
      if (dval < bestv[r]) { bestv[r] = dval; besti[r] = nCol; }
    }
  }

#pragma unroll
  for (int r = 0; r < 8; ++r) {
#pragma unroll
    for (int off = 8; off > 0; off >>= 1) {
      float ov = __shfl_xor(bestv[r], off, 16);
      int   oi = __shfl_xor(besti[r], off, 16);
      if (ov < bestv[r] || (ov == bestv[r] && oi < besti[r])) {
        bestv[r] = ov; besti[r] = oi;
      }
    }
  }

#pragma unroll
  for (int mloc = 0; mloc < 16; ++mloc) {
    int src  = (mloc < 8) ? 0 : 16;
    int code = __shfl(besti[mloc & 7], src, 32);
    const unsigned* ec = (const unsigned*)(cbh + code * 64);
    unsigned* qp = (unsigned*)(q + (size_t)(m_base + mloc) * 64);
    qp[lane] = ec[lane];            // 32 lanes x 2 halves = 64 contiguous
  }
}

// ---------------------------------------------------------------------------
// Repack w_t1 (in=64,out=32,4,4) f32 -> f16, 4 parity classes, tap-major:
// wp[par][oc][(kyi*2+kxi)*64 + ic],  ky=(py?2:3)-2*kyi, kx=(px?2:3)-2*kxi.
// ---------------------------------------------------------------------------
__global__ void repack_t1_kernel(const float* __restrict__ w,
                                 _Float16* __restrict__ wp) {
  int i = blockIdx.x * 256 + threadIdx.x;      // 32768
  int k = i & 255; int oc = (i >> 8) & 31; int par = i >> 13;
  int kyx = k >> 6, ic = k & 63;
  int kyi = kyx >> 1, kxi = kyx & 1;
  int ky = ((par >> 1) ? 2 : 3) - 2 * kyi;
  int kx = ((par & 1) ? 2 : 3) - 2 * kxi;
  wp[i] = (_Float16)w[((ic * 32 + oc) * 4 + ky) * 4 + kx];
}

// ---------------------------------------------------------------------------
// ConvTranspose2d t1 as WMMA per output-parity class (f16 NHWC in/out):
// M = 131072, N = 32, K = 256; bias + ReLU at store.
// ---------------------------------------------------------------------------
__global__ __launch_bounds__(128) void convt1_wmma_kernel(
    const _Float16* __restrict__ x,   // NHWC (32,64,64,64)
    const _Float16* __restrict__ wp, const float* __restrict__ bias,
    _Float16* __restrict__ out) {     // NHWC (32,128,128,32)
  constexpr int MREP = 2;
  const int lane = threadIdx.x, col = lane & 15, hi = lane >> 4;
  const int par = blockIdx.z;
  const int py = par >> 1, px = par & 1;
  const int offy = py - 1, offx = px - 1;
  const int job = blockIdx.x * blockDim.y + threadIdx.y;
  const int n = blockIdx.y * 16 + col;
  const _Float16* wn = wp + (size_t)(par * 32 + n) * 256 + hi * 8;

  int yq[MREP], xq[MREP], bOff[MREP];
#pragma unroll
  for (int r = 0; r < MREP; ++r) {
    int m = (job * MREP + r) * 16 + col;
    int bb = m >> 12, rem = m & 4095;
    yq[r] = rem >> 6; xq[r] = rem & 63;
    bOff[r] = bb << 12;
  }

  v8f acc[MREP];
#pragma unroll
  for (int r = 0; r < MREP; ++r) acc[r] = (v8f){};

#pragma unroll
  for (int step = 0; step < 8; ++step) {       // (kyi,kxi) x 2 ic-halves
    const int kyx = step >> 1, s = step & 1;
    const int kyi = kyx >> 1, kxi = kyx & 1;
    v16h bv = load_a_runs<false>(wn + step * 32);
#pragma unroll
    for (int r = 0; r < MREP; ++r) {
      int iy = yq[r] + offy + kyi, ix = xq[r] + offx + kxi;
      v16h av;
      if (iy >= 0 && iy < 64 && ix >= 0 && ix < 64) {
        const _Float16* ap =
            x + (size_t)(bOff[r] + (iy << 6) + ix) * 64 + s * 32 + hi * 8;
        av = load_a_runs<false>(ap);
      } else {
#pragma unroll
        for (int h = 0; h < 16; ++h) av[h] = (_Float16)0.f;
      }
      acc[r] = __builtin_amdgcn_wmma_f32_16x16x32_f16(
          false, av, false, bv, (short)0, acc[r], false, false);
    }
  }

  const float bn = bias[n];
#pragma unroll
  for (int r = 0; r < MREP; ++r) {
#pragma unroll
    for (int rr = 0; rr < 8; ++rr) {
      int ms = (job * MREP + r) * 16 + rr + 8 * hi;
      int bs = ms >> 12, rem = ms & 4095;
      int oy = ((rem >> 6) << 1) + py, ox = ((rem & 63) << 1) + px;
      size_t oidx = ((size_t)((bs << 14) + (oy << 7) + ox) << 5) + n;
      out[oidx] = (_Float16)fmaxf(acc[r][rr] + bn, 0.f);
    }
  }
}

// ---------------------------------------------------------------------------
// ConvTranspose2d t2: f16 NHWC (32,128,128,32) -> f32 (32,1,256,256).
// Contiguous 32-channel dot per tap; weights uniform (s_load).
// ---------------------------------------------------------------------------
__global__ void convt2_kernel(const _Float16* __restrict__ x,
                              const float* __restrict__ w,   // (32,1,4,4)
                              const float* __restrict__ bias,
                              float* __restrict__ out) {
  int idx = blockIdx.x * 256 + threadIdx.x;   // 2097152
  int ox = idx & 255; int t = idx >> 8;
  int oy = t & 255;   int b = t >> 8;
  float s = bias[0];
  const _Float16* xb = x + ((size_t)b << 19);
#pragma unroll
  for (int ky = 0; ky < 4; ++ky) {
    int ty = oy + 1 - ky;
    if (ty & 1) continue;
    int iy = ty >> 1;
    if (iy < 0 || iy >= 128) continue;
#pragma unroll
    for (int kx = 0; kx < 4; ++kx) {
      int tx = ox + 1 - kx;
      if (tx & 1) continue;
      int ix = tx >> 1;
      if (ix < 0 || ix >= 128) continue;
      const v8h* xp = (const v8h*)(xb + (size_t)(iy * 128 + ix) * 32);
      int kyx = ky * 4 + kx;
#pragma unroll
      for (int j = 0; j < 4; ++j) {
        v8h v = xp[j];
#pragma unroll
        for (int e = 0; e < 8; ++e)
          s += (float)v[e] * w[(8*j+e) * 16 + kyx];
      }
    }
  }
  out[idx] = s;
}

// ---------------------------------------------------------------------------
extern "C" void kernel_launch(void* const* d_in, const int* in_sizes, int n_in,
                              void* d_out, int out_size, void* d_ws, size_t ws_size,
                              hipStream_t stream) {
  const float* x        = (const float*)d_in[0];
  const float* w_e1     = (const float*)d_in[1];
  const float* b_e1     = (const float*)d_in[2];
  const float* w_e2     = (const float*)d_in[3];
  const float* b_e2     = (const float*)d_in[4];
  const float* w_er1a   = (const float*)d_in[5];
  const float* w_er1b   = (const float*)d_in[6];
  const float* w_er2a   = (const float*)d_in[7];
  const float* w_er2b   = (const float*)d_in[8];
  const float* w_pre    = (const float*)d_in[9];
  const float* b_pre    = (const float*)d_in[10];
  const float* codebook = (const float*)d_in[11];
  const float* w_d1     = (const float*)d_in[12];
  const float* b_d1     = (const float*)d_in[13];
  const float* w_dr1a   = (const float*)d_in[14];
  const float* w_dr1b   = (const float*)d_in[15];
  const float* w_dr2a   = (const float*)d_in[16];
  const float* w_dr2b   = (const float*)d_in[17];
  const float* w_t1     = (const float*)d_in[18];
  const float* b_t1     = (const float*)d_in[19];
  const float* w_t2     = (const float*)d_in[20];
  const float* b_t2     = (const float*)d_in[21];
  float* out = (float*)d_out;

  // Workspace layout (f16 activations, f32 norms); all written before read.
  _Float16* hw  = (_Float16*)d_ws;
  _Float16* bufH = hw;                    // 16777216 h : (32,128,128,32)
  _Float16* big0 = bufH + 16777216;       //  8388608 h : (32,64,64,64)
  _Float16* big1 = big0 + 8388608;
  _Float16* big2 = big1 + 8388608;
  _Float16* bufR = big2 + 8388608;        //  4194304 h : (32,64,64,32)
  _Float16* cbh  = bufR + 4194304;        //    32768 h : f16 codebook
  _Float16* wpT1 = cbh  + 32768;          //    32768 h
  _Float16* wpE2 = wpT1 + 32768;          //    32768 h : 64 x 512
  _Float16* wpA0 = wpE2 + 32768;          //    18432 h : er1a
  _Float16* wpA1 = wpA0 + 18432;          //    er2a
  _Float16* wpA2 = wpA1 + 18432;          //    dr1a
  _Float16* wpA3 = wpA2 + 18432;          //    dr2a
  _Float16* wpD1 = wpA3 + 18432;          //    36864 h : d1
  _Float16* wpR1B = wpD1 + 36864;         //     2048 h : er1b (64x32)
  _Float16* wpR2B = wpR1B + 2048;
  _Float16* wpD1B = wpR2B + 2048;         //    dr1b
  _Float16* wpD2B = wpD1B + 2048;         //    dr2b
  _Float16* wpPre = wpD2B + 2048;         //     4096 h : pre (64x64)
  float*    cbn   = (float*)(wpPre + 4096);   //  512 f32

  const dim3 wb(32, 4);                   // 4 waves/block, MREP=2 tiles/wave
  const int  BX = 131072 / 16 / 2 / 4;    // 1024 blocks.x for 64x64 maps

  // Weight prep + codebook (independent).
  repack_t1_kernel<<<128, 256, 0, stream>>>(w_t1, wpT1);
  repack_oihw_kernel<32,4,4><<<128, 256, 0, stream>>>(w_e2, wpE2, 32768);
  repack_oihw_kernel<64,3,3><<<72,  256, 0, stream>>>(w_er1a, wpA0, 18432);
  repack_oihw_kernel<64,3,3><<<72,  256, 0, stream>>>(w_er2a, wpA1, 18432);
  repack_oihw_kernel<64,3,3><<<72,  256, 0, stream>>>(w_dr1a, wpA2, 18432);
  repack_oihw_kernel<64,3,3><<<72,  256, 0, stream>>>(w_dr2a, wpA3, 18432);
  repack_oihw_kernel<64,3,3><<<144, 256, 0, stream>>>(w_d1,  wpD1, 36864);
  repack_oihw_kernel<32,1,1><<<8,   256, 0, stream>>>(w_er1b, wpR1B, 2048);
  repack_oihw_kernel<32,1,1><<<8,   256, 0, stream>>>(w_er2b, wpR2B, 2048);
  repack_oihw_kernel<32,1,1><<<8,   256, 0, stream>>>(w_dr1b, wpD1B, 2048);
  repack_oihw_kernel<32,1,1><<<8,   256, 0, stream>>>(w_dr2b, wpD2B, 2048);
  repack_oihw_kernel<64,1,1><<<16,  256, 0, stream>>>(w_pre, wpPre, 4096);
  repack_oihw_kernel<64,1,1><<<128, 256, 0, stream>>>(codebook, cbh, 32768);
  cbnorm_kernel<<<2, 256, 0, stream>>>(codebook, cbn);

  // Encoder
  conv_e1_kernel<<<2048, 256, 0, stream>>>(x, w_e1, b_e1, bufH);
  wmma_conv_kernel<32,128,128,64, 4,4,2,1, false,false,true,false>
      <<<dim3(BX, 4), wb, 0, stream>>>(bufH, wpE2, b_e2, nullptr, big0);
  // res block 1: 3x3 reads relu(h2) on the fly, stores relu'd; 1x1 adds resid
  wmma_conv_kernel<64,64,64,32, 3,3,1,1, true,true,false,false>
      <<<dim3(BX, 2), wb, 0, stream>>>(big0, wpA0, nullptr, nullptr, bufR);
  wmma_conv_kernel<32,64,64,64, 1,1,1,0, false,false,false,true>
      <<<dim3(BX, 4), wb, 0, stream>>>(bufR, wpR1B, nullptr, big0, big1);
  // res block 2
  wmma_conv_kernel<64,64,64,32, 3,3,1,1, true,true,false,false>
      <<<dim3(BX, 2), wb, 0, stream>>>(big1, wpA1, nullptr, nullptr, bufR);
  wmma_conv_kernel<32,64,64,64, 1,1,1,0, false,false,false,true>
      <<<dim3(BX, 4), wb, 0, stream>>>(bufR, wpR2B, nullptr, big1, big2);
  // pre-VQ 1x1 + bias
  wmma_conv_kernel<64,64,64,64, 1,1,1,0, false,false,true,false>
      <<<dim3(BX, 4), wb, 0, stream>>>(big2, wpPre, b_pre, nullptr, big0);

  // Vector quantization
  vq_kernel<<<2048, wb, 0, stream>>>(big0, cbh, cbn, big1);

  // Decoder
  wmma_conv_kernel<64,64,64,64, 3,3,1,1, false,false,true,false>
      <<<dim3(BX, 4), wb, 0, stream>>>(big1, wpD1, b_d1, nullptr, big0);
  wmma_conv_kernel<64,64,64,32, 3,3,1,1, true,true,false,false>
      <<<dim3(BX, 2), wb, 0, stream>>>(big0, wpA2, nullptr, nullptr, bufR);
  wmma_conv_kernel<32,64,64,64, 1,1,1,0, false,false,false,true>
      <<<dim3(BX, 4), wb, 0, stream>>>(bufR, wpD1B, nullptr, big0, big2);
  wmma_conv_kernel<64,64,64,32, 3,3,1,1, true,true,false,false>
      <<<dim3(BX, 2), wb, 0, stream>>>(big2, wpA3, nullptr, nullptr, bufR);
  wmma_conv_kernel<32,64,64,64, 1,1,1,0, false,false,false,true>
      <<<dim3(BX, 4), wb, 0, stream>>>(bufR, wpD2B, nullptr, big2, big1);

  // Transposed convs: t1 on WMMA (4 parity classes), t2 direct.
  convt1_wmma_kernel<<<dim3(BX, 2, 4), wb, 0, stream>>>(big1, wpT1, b_t1, bufH);
  convt2_kernel<<<8192, 256, 0, stream>>>(bufH, w_t2, b_t2, out);

  (void)in_sizes; (void)n_in; (void)out_size; (void)ws_size;
}
